// BayesianAttention_84181359002165
// MI455X (gfx1250) — matrix operand order repaired and checked by reading, not compiled, not run on hardware
//
#include <hip/hip_runtime.h>
#include <hip/hip_bf16.h>

// ---------------------------------------------------------------------------
// BayesianAttention on MI455X (gfx1250, wave32, WMMA, async-to-LDS).
// Precision: bf16 operands / f32 accumulate via v_wmma_f32_16x16x32_bf16.
// Pipeline:
//   [cvt x -> bf16] [transpose w* -> bf16 W^T] [prior-bias table]
//   -> [QKV gemm: DOUBLE-BUFFERED async LDS staging, fused over gridDim.z]
//   -> [flash attn + table bias]  -> [O gemm -> f32 d_out]
// GEMM overlaps global->LDS tile DMA (ASYNCcnt, in-order completion =>
// thresholded s_wait_asynccnt) with WMMA on the other buffer.
// ---------------------------------------------------------------------------

typedef __bf16 bf16_t;
typedef __attribute__((ext_vector_type(8)))  __bf16 v8bf;
typedef __attribute__((ext_vector_type(16))) __bf16 v16bf;
typedef __attribute__((ext_vector_type(8)))  float  v8f;

#define DIM 2048
#define SEQ 2048
#define NH  16
#define HD  128

static __device__ __forceinline__ v16bf cat16(v8bf lo, v8bf hi) {
  union { v16bf v; v8bf h[2]; } u; u.h[0] = lo; u.h[1] = hi; return u.v;
}

static __device__ __forceinline__ v8f wmma_bf16(v16bf a, v16bf b, v8f c) {
  // (neg_a, A, neg_b, B, c_mod, C, reuse_a, reuse_b)
  return __builtin_amdgcn_wmma_f32_16x16x32_bf16(false, a, false, b, (short)0, c,
                                                 false, false);
}

// CDNA5 async copy: 16B global -> LDS per lane, tracked by ASYNCcnt.
// Generic LDS pointers carry the LDS byte offset in their low 32 bits.
static __device__ __forceinline__ void async_g2l_b128(uint32_t lds_addr,
                                                      const bf16_t* g) {
  asm volatile("global_load_async_to_lds_b128 %0, %1, off"
               :: "v"(lds_addr), "v"((uint64_t)(uintptr_t)g)
               : "memory");
}
// Async loads complete in order: waiting for <=3 outstanding after issuing the
// next chunk's 3 transfers guarantees the previous chunk has fully landed.
static __device__ __forceinline__ void wait_async0() {
  asm volatile("s_wait_asynccnt 0" ::: "memory");
}
static __device__ __forceinline__ void wait_async3() {
  asm volatile("s_wait_asynccnt 3" ::: "memory");
}

// ---------------------------------------------------------------------------
// Element-wise f32 -> bf16 (for x).
// ---------------------------------------------------------------------------
__global__ __launch_bounds__(256) void cvt_bf16(const float* __restrict__ src,
                                                bf16_t* __restrict__ dst) {
  const size_t i = ((size_t)blockIdx.x * 256 + threadIdx.x) * 4;
  const float4 v = *(const float4*)(src + i);
  dst[i + 0] = (bf16_t)v.x;
  dst[i + 1] = (bf16_t)v.y;
  dst[i + 2] = (bf16_t)v.z;
  dst[i + 3] = (bf16_t)v.w;
}

// ---------------------------------------------------------------------------
// f32 [k][n] -> bf16 [n][k] (weights pre-transposed so GEMM B tiles and
// B fragments are contiguous). 32x32 tiles bounced through LDS.
// ---------------------------------------------------------------------------
__global__ __launch_bounds__(256) void cvt_transpose(const float* __restrict__ src,
                                                     bf16_t* __restrict__ dst) {
  __shared__ float t[32][33];
  const int nb = blockIdx.x * 32;            // n tile base
  const int kb = blockIdx.y * 32;            // k tile base
  const int tx = threadIdx.x & 31;
  const int ty = threadIdx.x >> 5;           // 8 rows per pass
  #pragma unroll
  for (int i = 0; i < 4; ++i)
    t[ty + i * 8][tx] = src[(size_t)(kb + ty + i * 8) * DIM + nb + tx];
  __syncthreads();
  #pragma unroll
  for (int i = 0; i < 4; ++i)
    dst[(size_t)(nb + ty + i * 8) * DIM + kb + tx] = (bf16_t)t[tx][ty + i * 8];
}

// ---------------------------------------------------------------------------
// Prior bias table: tab[h][d] = -(|(-d - loc_t)*e^scale| + eps)^shape,
// d = q_pos - k_pos in [0, SEQ). Replaces per-score transcendentals.
// ---------------------------------------------------------------------------
__global__ __launch_bounds__(256) void bias_table(const float* __restrict__ shape_p,
                                                  const float* __restrict__ scale_p,
                                                  const float* __restrict__ loc_p,
                                                  float* __restrict__ tab) {
  const int h = blockIdx.y;
  const int d = blockIdx.x * 256 + threadIdx.x;
  const float shp   = shape_p[h];
  const float es    = __expf(scale_p[h]);
  const float lc    = loc_p[h];
  const float loc_t = __expf(lc) - __expf(-lc);
  const float dist  = -(float)d;
  const float z     = (dist - loc_t) * es;
  tab[h * SEQ + d]  = -__powf(fabsf(z) + 1e-5f, shp);
}

// ---------------------------------------------------------------------------
// GEMM: C[z] = A (MxK bf16, row major) * W[z] (bf16, PRE-TRANSPOSED [n][k]).
// Block = 64(M) x 128(N), 256 threads (8 waves): wave tile 32x32 -> 4 WMMA
// per 32-deep K chunk. Double-buffered LDS; tiles DMA'd in with
// global_load_async_to_lds_b128 while the other buffer feeds the WMMAs.
// If z == transZ the output is written transposed (used for V -> V^T).
// ---------------------------------------------------------------------------
template <typename TC>
__global__ __launch_bounds__(256) void gemm_wmma(
    const bf16_t* __restrict__ A,
    const bf16_t* __restrict__ W0, const bf16_t* __restrict__ W1,
    const bf16_t* __restrict__ W2,
    TC* __restrict__ C0, TC* __restrict__ C1, TC* __restrict__ C2,
    int transZ)
{
  __shared__ __align__(16) bf16_t As[2][64][40];    // [buf][m][k], 80B rows
  __shared__ __align__(16) bf16_t Bs[2][128][40];   // [buf][n][k]
  const uint32_t ABUF = 64 * 40 * 2;                // 5120 B per A buffer
  const uint32_t BBUF = 128 * 40 * 2;               // 10240 B per B buffer

  const int z = blockIdx.z;
  const bf16_t* W = (z == 0) ? W0 : ((z == 1) ? W1 : W2);
  TC*           C = (z == 0) ? C0 : ((z == 1) ? C1 : C2);

  const int bm   = blockIdx.x * 64;
  const int bn   = blockIdx.y * 128;
  const int tid  = threadIdx.x;
  const int lane = tid & 31;
  const int wave = tid >> 5;
  const int lr   = lane & 15;
  const int hi   = lane >> 4;
  const int wm   = wave & 1;            // 0..1 (M)
  const int wn   = wave >> 1;           // 0..3 (N)
  const int kb   = hi ? 8 : 0;          // A-frag K striping
  const int kb2  = hi ? 16 : 0;         // unused (B frags use kb)

  const uint32_t ldsA = (uint32_t)(uintptr_t)&As[0][0][0];
  const uint32_t ldsB = (uint32_t)(uintptr_t)&Bs[0][0][0];
  const int aRow = tid >> 2, aSeg = tid & 3;        // A: 64 rows x 4 x 16B

  // 3 async instructions per wave per chunk (1 A segment + 2 B segments).
  auto issue = [&](int k0, int buf) {
    async_g2l_b128(ldsA + (uint32_t)buf * ABUF + (uint32_t)(aRow * 80 + aSeg * 16),
                   A + (size_t)(bm + aRow) * DIM + k0 + aSeg * 8);
    #pragma unroll
    for (int i = 0; i < 2; ++i) {                   // B: 128 rows x 4 x 16B
      const int e = tid + i * 256;
      const int r = e >> 2, s = e & 3;
      async_g2l_b128(ldsB + (uint32_t)buf * BBUF + (uint32_t)(r * 80 + s * 16),
                     W + (size_t)(bn + r) * DIM + k0 + s * 8);
    }
  };

  v8f acc[2][2];
  #pragma unroll
  for (int si = 0; si < 2; ++si)
    #pragma unroll
    for (int sj = 0; sj < 2; ++sj)
      #pragma unroll
      for (int r = 0; r < 8; ++r) acc[si][sj][r] = 0.f;

  const int NCH = DIM / 32;
  issue(0, 0);                                      // prologue
  for (int kc = 0; kc < NCH; ++kc) {
    const int cur = kc & 1;
    if (kc + 1 < NCH) { issue((kc + 1) * 32, cur ^ 1); wait_async3(); }
    else              { wait_async0(); }
    __syncthreads();                                // chunk kc visible to all

    v16bf a0, a1, b0, b1;
    {
      const int r0 = wm * 32 + lr;
      a0 = cat16(*(const v8bf*)&As[cur][r0][kb],
                 *(const v8bf*)&As[cur][r0][16 + kb]);
      a1 = cat16(*(const v8bf*)&As[cur][r0 + 16][kb],
                 *(const v8bf*)&As[cur][r0 + 16][16 + kb]);
      const int n0 = wn * 32 + lr;
      b0 = cat16(*(const v8bf*)&Bs[cur][n0][kb],
                 *(const v8bf*)&Bs[cur][n0][16 + kb]);
      b1 = cat16(*(const v8bf*)&Bs[cur][n0 + 16][kb],
                 *(const v8bf*)&Bs[cur][n0 + 16][16 + kb]);
    }
    acc[0][0] = wmma_bf16(a0, b0, acc[0][0]);
    acc[0][1] = wmma_bf16(a0, b1, acc[0][1]);
    acc[1][0] = wmma_bf16(a1, b0, acc[1][0]);
    acc[1][1] = wmma_bf16(a1, b1, acc[1][1]);
    __syncthreads();                                // done reading buffer cur
  }

  #pragma unroll
  for (int si = 0; si < 2; ++si)
    #pragma unroll
    for (int sj = 0; sj < 2; ++sj)
      #pragma unroll
      for (int r = 0; r < 8; ++r) {
        const int gr = bm + wm * 32 + si * 16 + r + 8 * hi;  // M = r + 8*half
        const int gc = bn + wn * 32 + sj * 16 + lr;          // N = lane%16
        if (z == transZ) C[(size_t)gc * DIM + gr] = (TC)acc[si][sj][r];
        else             C[(size_t)gr * DIM + gc] = (TC)acc[si][sj][r];
      }
}

// ---------------------------------------------------------------------------
// Flash attention with table-lookup prior bias. One wave per (q tile, head).
// Q fragments hoisted out of the key loop; K/V fragments are contiguous
// 16B global loads feeding WMMA; P bounced through LDS (D -> A layout).
// ---------------------------------------------------------------------------
__global__ __launch_bounds__(32) void attn_wmma(
    const bf16_t* __restrict__ Q, const bf16_t* __restrict__ K,
    const bf16_t* __restrict__ Vt, bf16_t* __restrict__ O,
    const float* __restrict__ tab)
{
  __shared__ __align__(16) bf16_t Ps[16][40];

  const int h    = blockIdx.y;
  const int qb   = blockIdx.x * 16;
  const int lane = threadIdx.x & 31;
  const int lr   = lane & 15;
  const int hi   = lane >> 4;
  const int kb   = hi ? 8 : 0;
  const int kb2  = hi ? 16 : 0;
  const float sqk = 0.088388347648318447f;      // 1/sqrt(128)
  const float* htab = tab + h * SEQ;

  v8f acc[8];
  #pragma unroll
  for (int j = 0; j < 8; ++j)
    #pragma unroll
    for (int r = 0; r < 8; ++r) acc[j][r] = 0.f;
  float m[8], l[8];
  #pragma unroll
  for (int r = 0; r < 8; ++r) { m[r] = -1e30f; l[r] = 0.f; }

  // Q fragments are key-loop invariant: load once.
  v16bf qf[4];
  {
    const bf16_t* qrow = Q + (size_t)(qb + lr) * DIM + h * HD;
    #pragma unroll
    for (int d0 = 0; d0 < 4; ++d0)
      qf[d0] = cat16(*(const v8bf*)(qrow + d0 * 32 + kb),
                     *(const v8bf*)(qrow + d0 * 32 + 16 + kb));
  }

  const int nch = (qb + 47) >> 5;               // chunks with kpos <= qb+15
  for (int kc = 0; kc < nch; ++kc) {
    const int kbase = kc * 32;
    v8f s0, s1;
    #pragma unroll
    for (int r = 0; r < 8; ++r) { s0[r] = 0.f; s1[r] = 0.f; }

    // ---- scores S = Q K^T over head_dim=128 ----
    #pragma unroll
    for (int d0 = 0; d0 < 4; ++d0) {
      const bf16_t* kr0 = K + (size_t)(kbase + lr) * DIM + h * HD + d0 * 32 + kb;
      const bf16_t* kr1 = kr0 + (size_t)16 * DIM;
      v16bf b0 = cat16(*(const v8bf*)kr0, *(const v8bf*)(kr0 + 16));
      v16bf b1 = cat16(*(const v8bf*)kr1, *(const v8bf*)(kr1 + 16));
      s0 = wmma_bf16(qf[d0], b0, s0);
      s1 = wmma_bf16(qf[d0], b1, s1);
    }

    // ---- scale + table bias + causal mask ----
    float sv[2][8];
    #pragma unroll
    for (int t = 0; t < 2; ++t)
      #pragma unroll
      for (int r = 0; r < 8; ++r) {
        const int qp_ = qb + r + 8 * hi;
        const int kp_ = kbase + t * 16 + lr;
        const int d   = qp_ - kp_;
        float v = (t ? s1[r] : s0[r]) * sqk;
        v = (d < 0) ? -1e9f : (v + htab[d]);
        sv[t][r] = v;
      }

    // ---- online softmax (row reduce across 16 lanes of each half) ----
    #pragma unroll
    for (int r = 0; r < 8; ++r) {
      float cm = fmaxf(sv[0][r], sv[1][r]);
      #pragma unroll
      for (int o = 1; o < 16; o <<= 1) cm = fmaxf(cm, __shfl_xor(cm, o, 32));
      const float mn    = fmaxf(m[r], cm);
      const float alpha = __expf(m[r] - mn);
      m[r] = mn;
      float rs = 0.f;
      #pragma unroll
      for (int t = 0; t < 2; ++t) {
        const float p = __expf(sv[t][r] - mn);
        sv[t][r] = p;
        rs += p;
      }
      #pragma unroll
      for (int o = 1; o < 16; o <<= 1) rs += __shfl_xor(rs, o, 32);
      l[r] = l[r] * alpha + rs;
      #pragma unroll
      for (int j = 0; j < 8; ++j) acc[j][r] *= alpha;
    }

    // ---- P: D layout -> LDS -> A layout ----
    __syncthreads();
    #pragma unroll
    for (int t = 0; t < 2; ++t)
      #pragma unroll
      for (int r = 0; r < 8; ++r)
        Ps[r + 8 * hi][t * 16 + lr] = (bf16_t)sv[t][r];
    __syncthreads();
    v16bf pf = cat16(*(const v8bf*)&Ps[lr][kb], *(const v8bf*)&Ps[lr][16 + kb]);

    // ---- O += P * V (V^T layout -> contiguous B fragments) ----
    #pragma unroll
    for (int j = 0; j < 8; ++j) {
      const bf16_t* vr = Vt + (size_t)(h * HD + j * 16 + lr) * SEQ + kbase + kb;
      v16bf vf = cat16(*(const v8bf*)vr, *(const v8bf*)(vr + 16));
      acc[j] = wmma_bf16(pf, vf, acc[j]);
    }
    __syncthreads();
  }

  #pragma unroll
  for (int r = 0; r < 8; ++r) {
    const float inv = 1.0f / l[r];
    const int row = qb + r + 8 * hi;
    #pragma unroll
    for (int j = 0; j < 8; ++j)
      O[(size_t)row * DIM + h * HD + j * 16 + lr] = (bf16_t)(acc[j][r] * inv);
  }
}

// ---------------------------------------------------------------------------
extern "C" void kernel_launch(void* const* d_in, const int* in_sizes, int n_in,
                              void* d_out, int out_size, void* d_ws, size_t ws_size,
                              hipStream_t stream) {
  const float* x       = (const float*)d_in[0];
  // d_in[1] = mask: causal, reproduced analytically
  const float* wq      = (const float*)d_in[2];
  const float* wk      = (const float*)d_in[3];
  const float* wv      = (const float*)d_in[4];
  const float* wo      = (const float*)d_in[5];
  const float* shape_p = (const float*)d_in[6];
  const float* scale_p = (const float*)d_in[7];
  const float* loc_p   = (const float*)d_in[8];
  // d_in[9] = start_pos == 0 (prefill)

  char* ws = (char*)d_ws;
  const size_t MAT = (size_t)DIM * DIM * sizeof(bf16_t);   // 8 MB each
  bf16_t* xb  = (bf16_t*)(ws + 0 * MAT);
  bf16_t* wqT = (bf16_t*)(ws + 1 * MAT);
  bf16_t* wkT = (bf16_t*)(ws + 2 * MAT);
  bf16_t* wvT = (bf16_t*)(ws + 3 * MAT);
  bf16_t* woT = (bf16_t*)(ws + 4 * MAT);
  bf16_t* Qb  = (bf16_t*)(ws + 5 * MAT);
  bf16_t* Kb  = (bf16_t*)(ws + 6 * MAT);
  bf16_t* Vt  = (bf16_t*)(ws + 7 * MAT);                   // transposed V
  bf16_t* Ab  = (bf16_t*)(ws + 8 * MAT);                   // attn out (pre-Wo)
  float*  tab = (float*)(ws + 9 * MAT);                    // 16*2048 f32 bias table
  float*  out = (float*)d_out;

  // One-shot precision/layout prep (bandwidth-trivial at 23.3 TB/s).
  cvt_bf16<<<dim3(DIM * DIM / 1024), 256, 0, stream>>>(x, xb);
  dim3 gt(DIM / 32, DIM / 32);
  cvt_transpose<<<gt, 256, 0, stream>>>(wq, wqT);
  cvt_transpose<<<gt, 256, 0, stream>>>(wk, wkT);
  cvt_transpose<<<gt, 256, 0, stream>>>(wv, wvT);
  cvt_transpose<<<gt, 256, 0, stream>>>(wo, woT);
  bias_table<<<dim3(SEQ / 256, NH), 256, 0, stream>>>(shape_p, scale_p, loc_p, tab);

  // Q/K/V projections (z selects weight); V written transposed.
  dim3 gqkv(DIM / 64, DIM / 128, 3);
  gemm_wmma<bf16_t><<<gqkv, 256, 0, stream>>>(
      xb, wqT, wkT, wvT, Qb, Kb, Vt, /*transZ=*/2);

  // Fused attention + prior + softmax.
  attn_wmma<<<dim3(SEQ / 16, NH), 32, 0, stream>>>(Qb, Kb, Vt, Ab, tab);

  // Output projection -> f32 result.
  dim3 go(DIM / 64, DIM / 128, 1);
  gemm_wmma<float><<<go, 256, 0, stream>>>(
      Ab, woT, woT, woT, out, out, out, /*transZ=*/-1);
}
